// SparseTransformer_43198781063554
// MI455X (gfx1250) — matrix operand order repaired and checked by reading, not compiled, not run on hardware
//
#include <hip/hip_runtime.h>
#include <hip/hip_bf16.h>

// Problem sizes (fixed by the reference)
#define B_  4096
#define N_  768
#define M_  16384
#define A_  256
#define EPS_ 1e-5f

#if defined(__has_builtin) && __has_builtin(__builtin_amdgcn_sched_barrier)
#define SCHED_FENCE() __builtin_amdgcn_sched_barrier(0)
#else
#define SCHED_FENCE()
#endif

typedef __bf16 bf16_t;
typedef __attribute__((ext_vector_type(16))) __bf16 v16bf;
typedef __attribute__((ext_vector_type(8)))  __bf16 v8bf;
typedef __attribute__((ext_vector_type(8)))  float  v8f;

// ---------------- fragment loaders (CDNA5 wave32 WMMA layouts, ISA 7.12.2) ----
// A (16x32 bf16): lane = row, runs [k0+half*8, +8) and [k0+16+half*8, +8)
__device__ __forceinline__ v16bf load_afrag_bf16(const bf16_t* p /* row + k0 + half*8 */) {
  v8bf lo = *(const v8bf*)p;
  v8bf hi = *(const v8bf*)(p + 16);
  v16bf f;
  #pragma unroll
  for (int i = 0; i < 8; ++i) { f[i] = lo[i]; f[i + 8] = hi[i]; }
  return f;
}
// B (32x16 bf16): lane = col, runs [k0+half*8, +8) and [k0+16+half*8, +8)
__device__ __forceinline__ v16bf load_bfrag_bf16(const bf16_t* p /* col row + k0 + half*8 */) {
  v8bf lo = *(const v8bf*)p;
  v8bf hi = *(const v8bf*)(p + 16);
  v16bf f;
  #pragma unroll
  for (int i = 0; i < 8; ++i) { f[i] = lo[i]; f[i + 8] = hi[i]; }
  return f;
}
// A fragment sourced from f32 (two 32B runs), converted to bf16 in registers
__device__ __forceinline__ v16bf load_afrag_f32(const float* p /* row + k0 + half*8 */) {
  float4 a0 = *(const float4*)(p);
  float4 a1 = *(const float4*)(p + 4);
  float4 a2 = *(const float4*)(p + 16);
  float4 a3 = *(const float4*)(p + 20);
  v16bf f;
  f[0]  = (bf16_t)a0.x; f[1]  = (bf16_t)a0.y; f[2]  = (bf16_t)a0.z; f[3]  = (bf16_t)a0.w;
  f[4]  = (bf16_t)a1.x; f[5]  = (bf16_t)a1.y; f[6]  = (bf16_t)a1.z; f[7]  = (bf16_t)a1.w;
  f[8]  = (bf16_t)a2.x; f[9]  = (bf16_t)a2.y; f[10] = (bf16_t)a2.z; f[11] = (bf16_t)a2.w;
  f[12] = (bf16_t)a3.x; f[13] = (bf16_t)a3.y; f[14] = (bf16_t)a3.z; f[15] = (bf16_t)a3.w;
  return f;
}

// ---------------- block reductions (256 threads) ----------------
__device__ __forceinline__ float block_sum_256(float v, float* red) {
  int t = threadIdx.x;
  red[t] = v; __syncthreads();
  #pragma unroll
  for (int s = 128; s > 0; s >>= 1) {
    if (t < s) red[t] += red[t + s];
    __syncthreads();
  }
  float r = red[0]; __syncthreads();
  return r;
}

// ---------------- LayerNorm of 256-wide rows (k branch) ----------------
__global__ void ln_rows_256(const float* __restrict__ W, const float* __restrict__ g,
                            const float* __restrict__ bb, bf16_t* __restrict__ out) {
  __shared__ float red[256];
  int m = blockIdx.x, t = threadIdx.x;
  float v  = W[(size_t)m * A_ + t];
  float mu = block_sum_256(v, red) * (1.0f / A_);
  float d  = v - mu;
  float var = block_sum_256(d * d, red) * (1.0f / A_);
  float y = d * rsqrtf(var + EPS_) * g[t] + bb[t];
  out[(size_t)m * A_ + t] = (bf16_t)y;
}

// ---------------- LayerNorm of 768-wide rows (v branch) ----------------
__global__ void ln_rows_768(const float* __restrict__ W, const float* __restrict__ g,
                            const float* __restrict__ bb, bf16_t* __restrict__ out) {
  __shared__ float red[256];
  int m = blockIdx.x, t = threadIdx.x;
  const float* row = W + (size_t)m * N_;
  float v0 = row[t], v1 = row[t + 256], v2 = row[t + 512];
  float mu = block_sum_256(v0 + v1 + v2, red) * (1.0f / N_);
  float d0 = v0 - mu, d1 = v1 - mu, d2 = v2 - mu;
  float var = block_sum_256(d0*d0 + d1*d1 + d2*d2, red) * (1.0f / N_);
  float rs = rsqrtf(var + EPS_);
  bf16_t* orow = out + (size_t)m * N_;
  orow[t]       = (bf16_t)(d0 * rs * g[t]       + bb[t]);
  orow[t + 256] = (bf16_t)(d1 * rs * g[t + 256] + bb[t + 256]);
  orow[t + 512] = (bf16_t)(d2 * rs * g[t + 512] + bb[t + 512]);
}

// ---------------- q = LN(x @ Wq^T + b) ; one block per batch row ----------------
__global__ void qproj_ln(const float* __restrict__ x, const float* __restrict__ Wq,
                         const float* __restrict__ qb, const float* __restrict__ g,
                         const float* __restrict__ bb, bf16_t* __restrict__ out) {
  __shared__ float xs[N_];
  __shared__ float red[256];
  int b = blockIdx.x, t = threadIdx.x;
  const float* xrow = x + (size_t)b * N_;
  for (int i = t; i < N_; i += 256) xs[i] = xrow[i];
  __syncthreads();
  const float* w = Wq + (size_t)t * N_;   // thread t streams its own weight row (L2-resident)
  float acc = qb[t];
  #pragma unroll 4
  for (int i = 0; i < N_; i += 4) {
    float4 wv = *(const float4*)(w + i);
    acc += xs[i] * wv.x + xs[i + 1] * wv.y + xs[i + 2] * wv.z + xs[i + 3] * wv.w;
  }
  float mu = block_sum_256(acc, red) * (1.0f / A_);
  float d  = acc - mu;
  float var = block_sum_256(d * d, red) * (1.0f / A_);
  float y = d * rsqrtf(var + EPS_) * g[t] + bb[t];
  out[(size_t)b * A_ + t] = (bf16_t)y;
}

// ---------------- LDS-tiled transpose: v_bf16[M,N] -> vT[N,M] ----------------
__global__ void transpose_bf16(const unsigned short* __restrict__ src,
                               unsigned short* __restrict__ dst) {
  __shared__ unsigned short tile[64][66];
  int m0 = blockIdx.x * 64, n0 = blockIdx.y * 64;
  int t = threadIdx.x;
  #pragma unroll
  for (int p = 0; p < 16; ++p) {
    int idx = t + p * 256; int i = idx >> 6, j = idx & 63;
    tile[i][j] = src[(size_t)(m0 + i) * N_ + n0 + j];       // coalesced read
  }
  __syncthreads();
  #pragma unroll
  for (int p = 0; p < 16; ++p) {
    int idx = t + p * 256; int i = idx >> 6, j = idx & 63;
    dst[(size_t)(n0 + i) * M_ + m0 + j] = tile[j][i];        // coalesced write
  }
}

// ---------------- S = q . k^T * (1/16)  via bf16 WMMA, software pipelined ----
// grid: (M/512, B/16), block 256 = 8 waves; each wave: one 16x64 tile (4 WMMA cols)
__global__ void scores_wmma(const bf16_t* __restrict__ qb, const bf16_t* __restrict__ kb,
                            float* __restrict__ S) {
  int lane = threadIdx.x & 31, w = threadIdx.x >> 5;
  int half = lane >> 4, n = lane & 15;
  int row0 = blockIdx.y * 16;
  int col0 = blockIdx.x * 512 + w * 64;

  const bf16_t* abase = qb + (size_t)(row0 + n) * A_ + half * 8;      // + kt*32
  const bf16_t* bbase0 = kb + (size_t)(col0 +  0 + n) * A_ + half * 8;
  const bf16_t* bbase1 = kb + (size_t)(col0 + 16 + n) * A_ + half * 8;
  const bf16_t* bbase2 = kb + (size_t)(col0 + 32 + n) * A_ + half * 8;
  const bf16_t* bbase3 = kb + (size_t)(col0 + 48 + n) * A_ + half * 8;

  v8f acc0 = {}, acc1 = {}, acc2 = {}, acc3 = {};

  // prologue: k-step 0
  v16bf a  = load_afrag_bf16(abase);
  v16bf b0 = load_bfrag_bf16(bbase0);
  v16bf b1 = load_bfrag_bf16(bbase1);
  v16bf b2 = load_bfrag_bf16(bbase2);
  v16bf b3 = load_bfrag_bf16(bbase3);

  #pragma unroll
  for (int kt = 0; kt < A_ / 32; ++kt) {
    v16bf an = {}, bn0 = {}, bn1 = {}, bn2 = {}, bn3 = {};
    const bool has_next = (kt + 1) < (A_ / 32);
    if (has_next) {                         // issue next k-step loads FIRST
      int k0 = (kt + 1) * 32;
      an  = load_afrag_bf16(abase  + k0);
      bn0 = load_bfrag_bf16(bbase0 + k0);
      bn1 = load_bfrag_bf16(bbase1 + k0);
      bn2 = load_bfrag_bf16(bbase2 + k0);
      bn3 = load_bfrag_bf16(bbase3 + k0);
    }
    SCHED_FENCE();                          // pin prefetch above the MMAs
    acc0 = __builtin_amdgcn_wmma_f32_16x16x32_bf16(false, a, false, b0, (short)0, acc0, false, false);
    acc1 = __builtin_amdgcn_wmma_f32_16x16x32_bf16(false, a, false, b1, (short)0, acc1, false, false);
    acc2 = __builtin_amdgcn_wmma_f32_16x16x32_bf16(false, a, false, b2, (short)0, acc2, false, false);
    acc3 = __builtin_amdgcn_wmma_f32_16x16x32_bf16(false, a, false, b3, (short)0, acc3, false, false);
    SCHED_FENCE();
    if (has_next) { a = an; b0 = bn0; b1 = bn1; b2 = bn2; b3 = bn3; }
  }

  int r0 = row0 + half * 8;
  #pragma unroll
  for (int r = 0; r < 8; ++r) {
    float* srow = S + (size_t)(r0 + r) * M_ + col0 + n;
    srow[0]  = acc0[r] * 0.0625f;
    srow[16] = acc1[r] * 0.0625f;
    srow[32] = acc2[r] * 0.0625f;
    srow[48] = acc3[r] * 0.0625f;
  }
}

// ---------------- in-place row softmax over M=16384 (row stays L2-hot) ----------------
__global__ void softmax_rows(float* __restrict__ S) {
  __shared__ float red[256];
  int b = blockIdx.x, t = threadIdx.x;
  float4* row = (float4*)(S + (size_t)b * M_);
  float mx = -3.402823466e38f;
  #pragma unroll
  for (int j = 0; j < 16; ++j) {
    float4 v = row[t + j * 256];
    mx = fmaxf(mx, fmaxf(fmaxf(v.x, v.y), fmaxf(v.z, v.w)));
  }
  red[t] = mx; __syncthreads();
  for (int s = 128; s > 0; s >>= 1) { if (t < s) red[t] = fmaxf(red[t], red[t + s]); __syncthreads(); }
  mx = red[0]; __syncthreads();
  float sum = 0.f;
  #pragma unroll
  for (int j = 0; j < 16; ++j) {
    float4 v = row[t + j * 256];
    v.x = __expf(v.x - mx); v.y = __expf(v.y - mx);
    v.z = __expf(v.z - mx); v.w = __expf(v.w - mx);
    row[t + j * 256] = v;
    sum += v.x + v.y + v.z + v.w;
  }
  red[t] = sum; __syncthreads();
  for (int s = 128; s > 0; s >>= 1) { if (t < s) red[t] += red[t + s]; __syncthreads(); }
  float inv = 1.0f / red[0];
  #pragma unroll
  for (int j = 0; j < 16; ++j) {
    float4 v = row[t + j * 256];
    v.x *= inv; v.y *= inv; v.z *= inv; v.w *= inv;
    row[t + j * 256] = v;
  }
}

// ---------------- x_hat = f . v  (K = 16384), pipelined, f cast f32->bf16 in regs ----
// grid: B/16 blocks, 8 waves; wave w covers cols [w*96, w*96+96) = 6 WMMA tiles
__global__ void xhat_wmma(const float* __restrict__ F, const bf16_t* __restrict__ vT,
                          float* __restrict__ out) {
  int lane = threadIdx.x & 31, w = threadIdx.x >> 5;
  int half = lane >> 4, n = lane & 15;
  int row0 = blockIdx.x * 16;
  int colbase = w * 96;

  const float*  abase  = F  + (size_t)(row0 + n) * M_ + half * 8;     // + kc
  const bf16_t* bbase0 = vT + (size_t)(colbase +  0 + n) * M_ + half * 8;
  const bf16_t* bbase1 = vT + (size_t)(colbase + 16 + n) * M_ + half * 8;
  const bf16_t* bbase2 = vT + (size_t)(colbase + 32 + n) * M_ + half * 8;
  const bf16_t* bbase3 = vT + (size_t)(colbase + 48 + n) * M_ + half * 8;
  const bf16_t* bbase4 = vT + (size_t)(colbase + 64 + n) * M_ + half * 8;
  const bf16_t* bbase5 = vT + (size_t)(colbase + 80 + n) * M_ + half * 8;

  v8f acc0 = {}, acc1 = {}, acc2 = {}, acc3 = {}, acc4 = {}, acc5 = {};

  v16bf a  = load_afrag_f32(abase);
  v16bf b0 = load_bfrag_bf16(bbase0);
  v16bf b1 = load_bfrag_bf16(bbase1);
  v16bf b2 = load_bfrag_bf16(bbase2);
  v16bf b3 = load_bfrag_bf16(bbase3);
  v16bf b4 = load_bfrag_bf16(bbase4);
  v16bf b5 = load_bfrag_bf16(bbase5);

  #pragma unroll 2
  for (int kc = 0; kc < M_; kc += 32) {
    v16bf an = {}, bn0 = {}, bn1 = {}, bn2 = {}, bn3 = {}, bn4 = {}, bn5 = {};
    const bool has_next = (kc + 32) < M_;
    if (has_next) {                         // next k-step loads before the MMAs
      int k1 = kc + 32;
      an  = load_afrag_f32 (abase  + k1);
      bn0 = load_bfrag_bf16(bbase0 + k1);
      bn1 = load_bfrag_bf16(bbase1 + k1);
      bn2 = load_bfrag_bf16(bbase2 + k1);
      bn3 = load_bfrag_bf16(bbase3 + k1);
      bn4 = load_bfrag_bf16(bbase4 + k1);
      bn5 = load_bfrag_bf16(bbase5 + k1);
    }
    SCHED_FENCE();                          // pin prefetch above the MMAs
    acc0 = __builtin_amdgcn_wmma_f32_16x16x32_bf16(false, a, false, b0, (short)0, acc0, false, false);
    acc1 = __builtin_amdgcn_wmma_f32_16x16x32_bf16(false, a, false, b1, (short)0, acc1, false, false);
    acc2 = __builtin_amdgcn_wmma_f32_16x16x32_bf16(false, a, false, b2, (short)0, acc2, false, false);
    acc3 = __builtin_amdgcn_wmma_f32_16x16x32_bf16(false, a, false, b3, (short)0, acc3, false, false);
    acc4 = __builtin_amdgcn_wmma_f32_16x16x32_bf16(false, a, false, b4, (short)0, acc4, false, false);
    acc5 = __builtin_amdgcn_wmma_f32_16x16x32_bf16(false, a, false, b5, (short)0, acc5, false, false);
    SCHED_FENCE();
    if (has_next) { a = an; b0 = bn0; b1 = bn1; b2 = bn2; b3 = bn3; b4 = bn4; b5 = bn5; }
  }

  int r0 = row0 + half * 8;
  #pragma unroll
  for (int r = 0; r < 8; ++r) {
    float* orow = out + (size_t)(r0 + r) * N_ + colbase + n;
    orow[0]  = acc0[r];
    orow[16] = acc1[r];
    orow[32] = acc2[r];
    orow[48] = acc3[r];
    orow[64] = acc4[r];
    orow[80] = acc5[r];
  }
}

extern "C" void kernel_launch(void* const* d_in, const int* in_sizes, int n_in,
                              void* d_out, int out_size, void* d_ws, size_t ws_size,
                              hipStream_t stream) {
  const float* x   = (const float*)d_in[0];
  const float* Wq  = (const float*)d_in[1];
  const float* Wqb = (const float*)d_in[2];
  const float* Wk  = (const float*)d_in[3];
  const float* Wv  = (const float*)d_in[4];
  const float* qg  = (const float*)d_in[5];
  const float* qbb = (const float*)d_in[6];
  const float* kg  = (const float*)d_in[7];
  const float* kbb = (const float*)d_in[8];
  const float* vg  = (const float*)d_in[9];
  const float* vbb = (const float*)d_in[10];

  // workspace layout (58 MB total, all 2MB-aligned offsets)
  char* ws = (char*)d_ws;
  bf16_t* q_bf = (bf16_t*)(ws + 0);                          //  2 MB : [B, A]
  bf16_t* k_bf = (bf16_t*)(ws + (size_t)2  * 1024 * 1024);   //  8 MB : [M, A]
  bf16_t* v_bf = (bf16_t*)(ws + (size_t)10 * 1024 * 1024);   // 24 MB : [M, N]
  bf16_t* vT   = (bf16_t*)(ws + (size_t)34 * 1024 * 1024);   // 24 MB : [N, M]

  float* xhat = (float*)d_out;                 // [B, N]
  float* Sf   = xhat + (size_t)B_ * N_;        // [B, M] : raw scores -> softmax in place

  ln_rows_256<<<M_, 256, 0, stream>>>(Wk, kg, kbb, k_bf);
  ln_rows_768<<<M_, 256, 0, stream>>>(Wv, vg, vbb, v_bf);
  qproj_ln<<<B_, 256, 0, stream>>>(x, Wq, Wqb, qg, qbb, q_bf);
  transpose_bf16<<<dim3(M_ / 64, N_ / 64), 256, 0, stream>>>(
      (const unsigned short*)v_bf, (unsigned short*)vT);
  scores_wmma<<<dim3(M_ / 512, B_ / 16), 256, 0, stream>>>(q_bf, k_bf, Sf);
  softmax_rows<<<B_, 256, 0, stream>>>(Sf);
  xhat_wmma<<<B_ / 16, 256, 0, stream>>>(Sf, vT, xhat);
}
